// CausalSelfAttention_26663156974104
// MI455X (gfx1250) — matrix operand order repaired and checked by reading, not compiled, not run on hardware
//
#include <hip/hip_runtime.h>
#include <hip/hip_bf16.h>
#include <stdint.h>

typedef _Float16 f16;
typedef __attribute__((ext_vector_type(16))) _Float16 v16h;
typedef __attribute__((ext_vector_type(8)))  float    v8f;

#define T_SEQ 2048
#define C_DIM 1024
#define N_HEAD 16
#define HD 64

// ---------------------------------------------------------------------------
// WMMA helpers (gfx1250, wave32).  D = A(16x32 f16) x B(32x16 f16) + C(16x16 f32)
// ---------------------------------------------------------------------------
__device__ __forceinline__ v8f wmma16(v16h a, v16h b, v8f c) {
  return __builtin_amdgcn_wmma_f32_16x16x32_f16(
      false, a, false, b, (short)0, c, false, false);
}

// CDNA5 async copy: global -> LDS, 16B per lane, tracked by ASYNCcnt (in-order).
__device__ __forceinline__ void async_load_b128(const f16* lds_dst, const f16* gaddr) {
  unsigned lds_off = (unsigned)(uintptr_t)lds_dst;   // generic LDS addr[31:0]
  asm volatile("global_load_async_to_lds_b128 %0, %1, off"
               :: "v"(lds_off), "v"(gaddr) : "memory");
}
template<int N>
__device__ __forceinline__ void wait_async() {
  asm volatile("s_wait_asynccnt %0" :: "i"(N) : "memory");
}

// A fragment: 16x32 (MxK), contiguous k = 16*half + s.
__device__ __forceinline__ v16h load_frag_a(const f16* base, int mbase, int stride, int lane) {
  const int half = lane >> 4;
  const f16* p = base + (mbase + (lane & 15)) * stride;
  v16h a;
#pragma unroll
  for (int v = 0; v < 8; ++v) {
    int k = half * 16 + v * 2;
    a[2 * v]     = p[k];
    a[2 * v + 1] = p[k + 1];
  }
  return a;
}

// B fragment: 32x16 (KxN), source stored [n][k].
__device__ __forceinline__ v16h load_frag_b(const f16* base, int nbase, int stride, int lane) {
  const int half = lane >> 4;
  const f16* p = base + (nbase + (lane & 15)) * stride;
  v16h b;
#pragma unroll
  for (int v = 0; v < 8; ++v) {
    int k = half * 16 + v * 2;
    b[2 * v]     = p[k];
    b[2 * v + 1] = p[k + 1];
  }
  return b;
}

// B fragment from a [k][n] (row-major KxN) LDS tile (V in P@V).
__device__ __forceinline__ v16h load_frag_b_kn(const f16* base, int nbase, int stride, int lane) {
  const int half = lane >> 4;
  const int n = nbase + (lane & 15);
  v16h b;
#pragma unroll
  for (int v = 0; v < 8; ++v) {
    int k = half * 16 + v * 2;
    b[2 * v]     = base[k * stride + n];
    b[2 * v + 1] = base[(k + 1) * stride + n];
  }
  return b;
}

__device__ __forceinline__ float half16_max(float v) {
  v = fmaxf(v, __shfl_xor(v, 1, 32));
  v = fmaxf(v, __shfl_xor(v, 2, 32));
  v = fmaxf(v, __shfl_xor(v, 4, 32));
  v = fmaxf(v, __shfl_xor(v, 8, 32));
  return v;
}
__device__ __forceinline__ float half16_sum(float v) {
  v += __shfl_xor(v, 1, 32);
  v += __shfl_xor(v, 2, 32);
  v += __shfl_xor(v, 4, 32);
  v += __shfl_xor(v, 8, 32);
  return v;
}

// ---------------------------------------------------------------------------
// f32 -> f16 conversion (4 elements / thread)
// ---------------------------------------------------------------------------
__global__ void cvt_f32_f16_kernel(const float* __restrict__ in, f16* __restrict__ out, int n4) {
  int i = blockIdx.x * blockDim.x + threadIdx.x;
  if (i < n4) {
    float4 v = ((const float4*)in)[i];
    out[4 * i + 0] = (f16)v.x;
    out[4 * i + 1] = (f16)v.y;
    out[4 * i + 2] = (f16)v.z;
    out[4 * i + 3] = (f16)v.w;
  }
}

// ---------------------------------------------------------------------------
// QKV GEMM: (B*T,3C) = X @ W^T + b, scattered to Q/K/V (B,H,T,hd) f16.
// Q is pre-scaled by sqrt(hd)=8 (folds the reference's score scaling).
// 256 thr (8 waves), WG tile 128x128, wave tile 32x64, K-step 32,
// double-buffered async global->LDS, last K-step peeled (branch-free body).
// ---------------------------------------------------------------------------
__global__ __launch_bounds__(256)
void qkv_gemm_kernel(const f16* __restrict__ X, const f16* __restrict__ W,
                     const float* __restrict__ bias,
                     f16* __restrict__ Q, f16* __restrict__ K, f16* __restrict__ V) {
  __shared__ f16 As[2][128 * 40];
  __shared__ f16 Bs[2][128 * 40];
  const int tid = threadIdx.x;
  const int lane = tid & 31, wave = tid >> 5;
  const int half = lane >> 4, ln = lane & 15;
  const int m0 = blockIdx.x * 128;
  const int n0 = blockIdx.y * 128;
  const int wr = (wave >> 1) * 32;
  const int wc = (wave & 1) * 64;

  // per-thread tile slice: 2 chunks of 16B per tile (128x32 halves / 256 thr)
  const int c0 = tid, c1 = 256 + tid;
  const int r0 = c0 >> 2, cc0 = (c0 & 3) * 8;
  const int r1 = c1 >> 2, cc1 = (c1 & 3) * 8;

  const v8f vzero = {0.f, 0.f, 0.f, 0.f, 0.f, 0.f, 0.f, 0.f};
  v8f acc[2][4];
#pragma unroll
  for (int mi = 0; mi < 2; ++mi)
#pragma unroll
    for (int ni = 0; ni < 4; ++ni) acc[mi][ni] = vzero;

  auto issue_tile = [&](int kk, int buf) {
    async_load_b128(&As[buf][r0 * 40 + cc0], &X[(m0 + r0) * C_DIM + kk + cc0]);
    async_load_b128(&Bs[buf][r0 * 40 + cc0], &W[(n0 + r0) * C_DIM + kk + cc0]);
    async_load_b128(&As[buf][r1 * 40 + cc1], &X[(m0 + r1) * C_DIM + kk + cc1]);
    async_load_b128(&Bs[buf][r1 * 40 + cc1], &W[(n0 + r1) * C_DIM + kk + cc1]);
  };
  auto compute = [&](int buf) {
    v16h a0 = load_frag_a(As[buf], wr,      40, lane);
    v16h a1 = load_frag_a(As[buf], wr + 16, 40, lane);
#pragma unroll
    for (int ni = 0; ni < 4; ++ni) {
      v16h b = load_frag_b(Bs[buf], wc + ni * 16, 40, lane);
      acc[0][ni] = wmma16(a0, b, acc[0][ni]);
      acc[1][ni] = wmma16(a1, b, acc[1][ni]);
    }
  };

  issue_tile(0, 0);
  for (int k0 = 0; k0 < C_DIM - 32; k0 += 32) {   // branch-free steady state
    const int buf = (k0 >> 5) & 1;
    issue_tile(k0 + 32, buf ^ 1);
    wait_async<4>();              // current tile landed (in-order completion)
    __syncthreads();
    compute(buf);
    __syncthreads();              // reads done before this buf is refilled
  }
  wait_async<0>();                // peeled last K-step
  __syncthreads();
  compute(((C_DIM - 32) >> 5) & 1);

  // epilogue: + bias (Q pre-scaled by 8), scatter to (B,H,T,hd)
  const int bb = m0 >> 11;
#pragma unroll
  for (int ni = 0; ni < 4; ++ni) {
    const int n   = n0 + wc + ni * 16 + ln;
    const int n64 = n >> 6;                 // 0..47
    const int h   = (n64 * 171) >> 9;       // n64 / 3 (exact for 0..511)
    const int si  = n64 - 3 * h;            // (n/64) % 3
    const int d   = n & (HD - 1);
    const float bn = bias[n];
    const float sc = (si == 0) ? 8.0f : 1.0f;   // sqrt(hd) folded into Q
    f16* dstp = ((si == 0) ? Q : (si == 1) ? K : V)
                + ((size_t)(bb * N_HEAD + h) * T_SEQ) * HD + d;
#pragma unroll
    for (int mi = 0; mi < 2; ++mi)
#pragma unroll
      for (int r = 0; r < 8; ++r) {
        int t = (m0 + wr + mi * 16 + r + 8 * half) & (T_SEQ - 1);
        dstp[(size_t)t * HD] = (f16)((acc[mi][ni][r] + bn) * sc);
      }
  }
}

// ---------------------------------------------------------------------------
// Flash attention: 64 query rows / WG (4 waves x 16 rows), 64-wide key blocks,
// double-buffered async K/V tiles; diagonal (masked) block peeled out of loop.
// ---------------------------------------------------------------------------
__global__ __launch_bounds__(128)
void attn_kernel(const f16* __restrict__ Q, const f16* __restrict__ K,
                 const f16* __restrict__ V, f16* __restrict__ O) {
  __shared__ f16 Qs[64 * 72];
  __shared__ f16 Ks[2][64 * 72];
  __shared__ f16 Vs[2][64 * 72];
  __shared__ f16 Ps[64 * 72];
  const int tid = threadIdx.x;
  const int lane = tid & 31, wave = tid >> 5;
  const int half = lane >> 4, ln = lane & 15;
  const int bh = blockIdx.x;                 // b*H + h
  const int qb = blockIdx.y;
  const int q0 = qb * 64;
  const f16* Qb = Q + (size_t)bh * T_SEQ * HD;
  const f16* Kb = K + (size_t)bh * T_SEQ * HD;
  const f16* Vb = V + (size_t)bh * T_SEQ * HD;

  // async Q tile (4 chunks/thr) + K/V tile 0 (8 chunks/thr)
#pragma unroll
  for (int i = 0; i < 4; ++i) {
    int c = i * 128 + tid;
    int r = c >> 3, cc = (c & 7) * 8;
    async_load_b128(&Qs[r * 72 + cc], &Qb[(q0 + r) * HD + cc]);
  }
#pragma unroll
  for (int i = 0; i < 4; ++i) {
    int c = i * 128 + tid;
    int r = c >> 3, cc = (c & 7) * 8;
    async_load_b128(&Ks[0][r * 72 + cc], &Kb[r * HD + cc]);
    async_load_b128(&Vs[0][r * 72 + cc], &Vb[r * HD + cc]);
  }

  const int qr = wave * 16;
  const v8f vzero = {0.f, 0.f, 0.f, 0.f, 0.f, 0.f, 0.f, 0.f};
  v8f o[4];
  float mrow[8], lrow[8];
#pragma unroll
  for (int dt = 0; dt < 4; ++dt) o[dt] = vzero;
#pragma unroll
  for (int r = 0; r < 8; ++r) { mrow[r] = -3.0e38f; lrow[r] = 0.f; }

  // one key block: S=QK^T, online softmax, O += P V.  diag => causal mask.
  auto process = [&](int k0, int buf, bool diag) {
    v8f s[4];
#pragma unroll
    for (int nt = 0; nt < 4; ++nt) s[nt] = vzero;
#pragma unroll
    for (int ks = 0; ks < 2; ++ks) {
      v16h a = load_frag_a(Qs + ks * 32, qr, 72, lane);
#pragma unroll
      for (int nt = 0; nt < 4; ++nt) {
        v16h bk = load_frag_b(&Ks[buf][ks * 32], nt * 16, 72, lane);
        s[nt] = wmma16(a, bk, s[nt]);
      }
    }

    if (diag) {          // causal mask, diagonal block only
#pragma unroll
      for (int nt = 0; nt < 4; ++nt)
#pragma unroll
        for (int r = 0; r < 8; ++r) {
          int kk = nt * 16 + ln;            // block-local
          int qq = qr + r + 8 * half;
          if (kk > qq) s[nt][r] = -3.0e38f;
        }
    }

#pragma unroll
    for (int r = 0; r < 8; ++r) {
      float mx = fmaxf(fmaxf(s[0][r], s[1][r]), fmaxf(s[2][r], s[3][r]));
      mx = half16_max(mx);
      float mnew = fmaxf(mrow[r], mx);
      float corr = __expf(mrow[r] - mnew);
      float rs = 0.f;
#pragma unroll
      for (int nt = 0; nt < 4; ++nt) {
        float p = __expf(s[nt][r] - mnew);
        s[nt][r] = p;
        rs += p;
      }
      rs = half16_sum(rs);
      lrow[r] = lrow[r] * corr + rs;
      mrow[r] = mnew;
#pragma unroll
      for (int dt = 0; dt < 4; ++dt) o[dt][r] *= corr;
    }

    // P -> LDS (f16) to re-shape into A-fragments
#pragma unroll
    for (int nt = 0; nt < 4; ++nt)
#pragma unroll
      for (int r = 0; r < 8; ++r)
        Ps[(qr + r + 8 * half) * 72 + nt * 16 + ln] = (f16)s[nt][r];
    __syncthreads();

#pragma unroll
    for (int ks = 0; ks < 2; ++ks) {
      v16h a = load_frag_a(Ps + ks * 32, qr, 72, lane);
#pragma unroll
      for (int dt = 0; dt < 4; ++dt) {
        v16h bv = load_frag_b_kn(&Vs[buf][ks * 32 * 72], dt * 16, 72, lane);
        o[dt] = wmma16(a, bv, o[dt]);
      }
    }
    __syncthreads();     // all waves done with this buf before it is refilled
  };

  for (int jb = 0; jb < qb; ++jb) {          // branch-free steady state
    const int k0 = jb * 64;
    const int buf = jb & 1;
#pragma unroll
    for (int i = 0; i < 4; ++i) {            // prefetch next K/V tiles
      int c = i * 128 + tid;
      int r = c >> 3, cc = (c & 7) * 8;
      async_load_b128(&Ks[buf ^ 1][r * 72 + cc], &Kb[(k0 + 64 + r) * HD + cc]);
      async_load_b128(&Vs[buf ^ 1][r * 72 + cc], &Vb[(k0 + 64 + r) * HD + cc]);
    }
    wait_async<8>();      // current K/V (and Q) landed
    __syncthreads();
    process(k0, buf, false);
  }
  wait_async<0>();                            // peeled diagonal block
  __syncthreads();
  process(q0, qb & 1, true);

  // normalize (reciprocal once per row) + store (B,T,C) f16
  const int bb = bh >> 4, h = bh & 15;
  float rinv[8];
#pragma unroll
  for (int r = 0; r < 8; ++r) rinv[r] = 1.0f / lrow[r];
#pragma unroll
  for (int dt = 0; dt < 4; ++dt)
#pragma unroll
    for (int r = 0; r < 8; ++r) {
      int qq = q0 + qr + r + 8 * half;
      int d  = dt * 16 + ln;
      O[((size_t)(bb * T_SEQ + qq)) * C_DIM + h * HD + d] = (f16)(o[dt][r] * rinv[r]);
    }
}

// ---------------------------------------------------------------------------
// Projection GEMM: out(B*T,C) = A @ Wp^T + b (f32 out), double-buffered,
// last K-step peeled.
// ---------------------------------------------------------------------------
__global__ __launch_bounds__(256)
void proj_gemm_kernel(const f16* __restrict__ A, const f16* __restrict__ W,
                      const float* __restrict__ bias, float* __restrict__ out) {
  __shared__ f16 As[2][128 * 40];
  __shared__ f16 Bs[2][128 * 40];
  const int tid = threadIdx.x;
  const int lane = tid & 31, wave = tid >> 5;
  const int half = lane >> 4, ln = lane & 15;
  const int m0 = blockIdx.x * 128;
  const int n0 = blockIdx.y * 128;
  const int wr = (wave >> 1) * 32;
  const int wc = (wave & 1) * 64;

  const int c0 = tid, c1 = 256 + tid;
  const int r0 = c0 >> 2, cc0 = (c0 & 3) * 8;
  const int r1 = c1 >> 2, cc1 = (c1 & 3) * 8;

  const v8f vzero = {0.f, 0.f, 0.f, 0.f, 0.f, 0.f, 0.f, 0.f};
  v8f acc[2][4];
#pragma unroll
  for (int mi = 0; mi < 2; ++mi)
#pragma unroll
    for (int ni = 0; ni < 4; ++ni) acc[mi][ni] = vzero;

  auto issue_tile = [&](int kk, int buf) {
    async_load_b128(&As[buf][r0 * 40 + cc0], &A[(m0 + r0) * C_DIM + kk + cc0]);
    async_load_b128(&Bs[buf][r0 * 40 + cc0], &W[(n0 + r0) * C_DIM + kk + cc0]);
    async_load_b128(&As[buf][r1 * 40 + cc1], &A[(m0 + r1) * C_DIM + kk + cc1]);
    async_load_b128(&Bs[buf][r1 * 40 + cc1], &W[(n0 + r1) * C_DIM + kk + cc1]);
  };
  auto compute = [&](int buf) {
    v16h a0 = load_frag_a(As[buf], wr,      40, lane);
    v16h a1 = load_frag_a(As[buf], wr + 16, 40, lane);
#pragma unroll
    for (int ni = 0; ni < 4; ++ni) {
      v16h b = load_frag_b(Bs[buf], wc + ni * 16, 40, lane);
      acc[0][ni] = wmma16(a0, b, acc[0][ni]);
      acc[1][ni] = wmma16(a1, b, acc[1][ni]);
    }
  };

  issue_tile(0, 0);
  for (int k0 = 0; k0 < C_DIM - 32; k0 += 32) {
    const int buf = (k0 >> 5) & 1;
    issue_tile(k0 + 32, buf ^ 1);
    wait_async<4>();
    __syncthreads();
    compute(buf);
    __syncthreads();
  }
  wait_async<0>();
  __syncthreads();
  compute(((C_DIM - 32) >> 5) & 1);

#pragma unroll
  for (int ni = 0; ni < 4; ++ni) {
    const int n = n0 + wc + ni * 16 + ln;
    const float bn = bias[n];
#pragma unroll
    for (int mi = 0; mi < 2; ++mi)
#pragma unroll
      for (int r = 0; r < 8; ++r) {
        int m = m0 + wr + mi * 16 + r + 8 * half;
        out[(size_t)m * C_DIM + n] = acc[mi][ni][r] + bn;
      }
  }
}

// ---------------------------------------------------------------------------
// Launch.  ws: Q 8MB | K 8MB | V 8MB | attnOut 8MB | x 8MB | qkv_w 6MB | proj_w 2MB
// ---------------------------------------------------------------------------
extern "C" void kernel_launch(void* const* d_in, const int* in_sizes, int n_in,
                              void* d_out, int out_size, void* d_ws, size_t ws_size,
                              hipStream_t stream) {
  const float* x      = (const float*)d_in[0];
  const float* qkv_w  = (const float*)d_in[1];
  const float* qkv_b  = (const float*)d_in[2];
  const float* proj_w = (const float*)d_in[3];
  const float* proj_b = (const float*)d_in[4];
  float* out = (float*)d_out;

  char* ws = (char*)d_ws;
  const size_t MB = 1024 * 1024;
  f16* Qh = (f16*)(ws + 0 * MB);
  f16* Kh = (f16*)(ws + 8 * MB);
  f16* Vh = (f16*)(ws + 16 * MB);
  f16* Ah = (f16*)(ws + 24 * MB);
  f16* Xh = (f16*)(ws + 32 * MB);
  f16* Wq = (f16*)(ws + 40 * MB);
  f16* Wp = (f16*)(ws + 46 * MB);

  cvt_f32_f16_kernel<<<4096, 256, 0, stream>>>(x, Xh, (2 * T_SEQ * C_DIM) / 4);
  cvt_f32_f16_kernel<<<3072, 256, 0, stream>>>(qkv_w, Wq, (3 * C_DIM * C_DIM) / 4);
  cvt_f32_f16_kernel<<<1024, 256, 0, stream>>>(proj_w, Wp, (C_DIM * C_DIM) / 4);

  qkv_gemm_kernel<<<dim3(32, 24), 256, 0, stream>>>(Xh, Wq, qkv_b, Qh, Kh, Vh);
  attn_kernel<<<dim3(32, 32), 128, 0, stream>>>(Qh, Kh, Vh, Ah);
  proj_gemm_kernel<<<dim3(32, 8), 256, 0, stream>>>(Ah, Wp, proj_b, out);
}
